// WaveNetPrior_2267742732372
// MI455X (gfx1250) — matrix operand order repaired and checked by reading, not compiled, not run on hardware
//
#include <hip/hip_runtime.h>

#define T_LEN   4096
#define NB      4
#define NC      256
#define NLAYERS 8
#define TT      64          // time tile per workgroup
#define XP      264         // padded LDS row pitch for 256-ch rows (bf16 elems)
#define SP      104         // padded LDS row pitch for spect rows (96+8)
#define SPK     96          // cond K padded 80 -> 96
#define NTHREADS 512        // 16 waves

typedef __attribute__((ext_vector_type(16))) __bf16 v16bf;
typedef __attribute__((ext_vector_type(8)))  __bf16 v8bf;
typedef __attribute__((ext_vector_type(8)))  float  v8f;
typedef __attribute__((ext_vector_type(4)))  float  v4f;

// packed-weight region sizes (elements)
#define NIN_PACK   (8*3*32*8*512)   // layer, tap, mtile, kstep, 512 lane-ordered bf16
#define NRES_PACK  (8*32*8*512)
#define NCOND_PACK (8*32*3*512)
#define NBIAS      (8*512)

static __device__ __forceinline__ v16bf lds_loadB(const __bf16* p) {
  // 32B per lane, 16B-aligned rows -> two 16B LDS reads
  union { v16bf v; v8bf h[2]; } u;
  u.h[0] = *(const v8bf*)(p);
  u.h[1] = *(const v8bf*)(p + 8);
  return u.v;
}

static __device__ __forceinline__ v8f wmma_bf16(v16bf a, v16bf b, v8f c) {
  return __builtin_amdgcn_wmma_f32_16x16x32_bf16(false, a, false, b, (short)0, c,
                                                 false, false);
}

// ---------------- weight pre-pack: f32 -> bf16 in WMMA A-operand lane order ---
// A 16x32 bf16 order used here (same as B reads): lane holds row M=lane&15;
// slot j: K = (lane>>4)*16 + j
__global__ void wn_prep_pack(const float* __restrict__ in_w,
                             const float* __restrict__ res_w,
                             const float* __restrict__ cond_w,
                             const float* __restrict__ in_b,
                             const float* __restrict__ cond_b,
                             __bf16* __restrict__ inP,
                             __bf16* __restrict__ resP,
                             __bf16* __restrict__ condP,
                             float* __restrict__ bias512) {
  long idx = (long)blockIdx.x * blockDim.x + threadIdx.x;
  if (idx < (long)NIN_PACK) {
    int e = (int)(idx & 511), lane = e >> 4, j = e & 15;
    long r = idx >> 9;                       // ((layer*3+tap)*32+mt)*8+ks
    int ks = (int)(r & 7), mt = (int)((r >> 3) & 31);
    int lt = (int)(r >> 8), tap = lt % 3, layer = lt / 3;
    int m = mt * 16 + (lane & 15);
    int K = ((lane >> 4) << 4) + j;
    int cin = ks * 32 + K;
    inP[idx] = (__bf16)in_w[(((long)layer * 512 + m) * 256 + cin) * 3 + tap];
    return;
  }
  idx -= NIN_PACK;
  if (idx < (long)NRES_PACK) {
    int e = (int)(idx & 511), lane = e >> 4, j = e & 15;
    long r = idx >> 9;                       // (layer*32+mt)*8+ks
    int ks = (int)(r & 7), mt = (int)((r >> 3) & 31), layer = (int)(r >> 8);
    int m = mt * 16 + (lane & 15);
    int K = ((lane >> 4) << 4) + j;
    int cin = ks * 32 + K;
    resP[idx] = (__bf16)res_w[((long)layer * 512 + m) * 256 + cin];
    return;
  }
  idx -= NRES_PACK;
  if (idx < (long)NCOND_PACK) {
    int e = (int)(idx & 511), lane = e >> 4, j = e & 15;
    long r = idx >> 9;                       // (layer*32+mt)*3+ks
    int ks = (int)(r % 3);
    long q = r / 3;
    int mt = (int)(q & 31), layer = (int)(q >> 5);
    int m = mt * 16 + (lane & 15);
    int K = ((lane >> 4) << 4) + j;
    int cin = ks * 32 + K;
    condP[idx] = (cin < 80) ? (__bf16)cond_w[((long)layer * 512 + m) * 80 + cin]
                            : (__bf16)0.0f;
    return;
  }
  idx -= NCOND_PACK;
  if (idx < (long)NBIAS) bias512[idx] = in_b[idx] + cond_b[idx];
}

// ---------------- start conv: 8 -> 256, 1x1 ---------------------------------
__global__ void wn_start(const float* __restrict__ audio,
                         const float* __restrict__ sw,
                         const float* __restrict__ sb,
                         float* __restrict__ h) {
  long idx = (long)blockIdx.x * blockDim.x + threadIdx.x;
  if (idx >= (long)NB * NC * T_LEN) return;
  int t = (int)(idx % T_LEN);
  int c = (int)((idx / T_LEN) & (NC - 1));
  int b = (int)(idx / ((long)NC * T_LEN));
  float acc = sb[c];
#pragma unroll
  for (int i = 0; i < 8; ++i)
    acc += sw[c * 8 + i] * audio[((long)b * 8 + i) * T_LEN + t];
  h[idx] = acc;
}

// ---------------- fused WN layer: dilated conv + cond + gate + res/skip -----
__global__ __launch_bounds__(NTHREADS) void wn_layer(
    const float* __restrict__ hIn, float* __restrict__ hOut,
    float* __restrict__ outAcc, const float* __restrict__ spect,
    const __bf16* __restrict__ inP, const __bf16* __restrict__ condP,
    const __bf16* __restrict__ resP, const float* __restrict__ bias512,
    const float* __restrict__ res_b, int layer, int dilation, int flags) {
  __shared__ __align__(32) char smem[3 * TT * XP * 2 + TT * SP * 2]; // 114,688 B
  __bf16* X = (__bf16*)smem;                     // [3][TT][XP] h halo tiles [n][c]
  __bf16* S = (__bf16*)(smem + 3 * TT * XP * 2); // [TT][SP] spect tile
  __bf16* A = (__bf16*)smem;                     // acts tile, reuses X after barrier

  const int tid = threadIdx.x;
  const int b = blockIdx.y;
  const int t0 = blockIdx.x * TT;

  // ---- stage h (3 taps, transposed, zero-padded) and spect into LDS ----
  {
    int c = tid & (NC - 1);       // 0..255
    int half = tid >> 8;          // 0/1 -> n range split
    int n0 = half * (TT / 2), n1 = n0 + (TT / 2);
    const float* hrow = hIn + ((long)b * NC + c) * T_LEN;
#pragma unroll
    for (int tap = 0; tap < 3; ++tap) {
      int tb = t0 + (tap - 1) * dilation;
      for (int n = n0; n < n1; ++n) {
        int t = tb + n;
        float v = (t >= 0 && t < T_LEN) ? hrow[t] : 0.0f;
        X[(tap * TT + n) * XP + c] = (__bf16)v;
      }
    }
    if (c < SPK && half == 0) {
      const float* srow = spect + ((long)b * 80 + c) * T_LEN;
      for (int n = 0; n < TT; ++n) {
        float v = (c < 80) ? srow[t0 + n] : 0.0f;
        S[n * SP + c] = (__bf16)v;
      }
    }
  }
  __syncthreads();

  const int lane = tid & 31;
  const int w = tid >> 5;        // 0..15 waves
  const int nt = w & 3;          // time sub-tile (4 x 16)
  const int g = w >> 2;          // m-group 0..3
  const int lo = lane & 15;
  const int hi = lane >> 4;
  const int nIdx = nt * 16 + lo; // local time row for B/C operands
  const int lane16 = lane * 16;

  // ---- in_act accumulators: 4 tanh tiles (m: 4g..4g+3) + 4 sigmoid tiles ----
  v8f accT[4], accS[4];
#pragma unroll
  for (int p = 0; p < 4; ++p) {
    int mT = (4 * g + p) * 16 + 8 * hi;      // multiple of 8 -> 32B aligned
    const v4f* bt = (const v4f*)(bias512 + layer * 512 + mT);
    const v4f* bs = (const v4f*)(bias512 + layer * 512 + 256 + mT);
    v4f t0v = bt[0], t1v = bt[1], s0v = bs[0], s1v = bs[1];
#pragma unroll
    for (int r = 0; r < 4; ++r) {
      accT[p][r] = t0v[r]; accT[p][r + 4] = t1v[r];
      accS[p][r] = s0v[r]; accS[p][r + 4] = s1v[r];
    }
  }

  // cond GEMM: 512x96 @ 96x16 (80 real + 16 zero K)
  {
    const __bf16* cbase = condP + ((long)layer * 32) * 3 * 512;
#pragma unroll
    for (int ks = 0; ks < 3; ++ks) {
      v16bf bop = lds_loadB(S + nIdx * SP + ks * 32 + hi * 16);
#pragma unroll
      for (int p = 0; p < 4; ++p) {
        v16bf aT = *(const v16bf*)(cbase + ((4 * g + p) * 3 + ks) * 512 + lane16);
        accT[p] = wmma_bf16(aT, bop, accT[p]);
        v16bf aS = *(const v16bf*)(cbase + ((16 + 4 * g + p) * 3 + ks) * 512 + lane16);
        accS[p] = wmma_bf16(aS, bop, accS[p]);
      }
    }
  }

  // dilated conv: 3 taps x (K=256 in 8 chunks of 32)
  for (int tap = 0; tap < 3; ++tap) {
    const __bf16* wbase = inP + (((long)layer * 3 + tap) * 32) * 8 * 512;
    const __bf16* xrow = X + (tap * TT + nIdx) * XP + hi * 16;
#pragma unroll
    for (int ks = 0; ks < 8; ++ks) {
      v16bf bop = lds_loadB(xrow + ks * 32);
#pragma unroll
      for (int p = 0; p < 4; ++p) {
        v16bf aT = *(const v16bf*)(wbase + ((4 * g + p) * 8 + ks) * 512 + lane16);
        accT[p] = wmma_bf16(aT, bop, accT[p]);
        v16bf aS = *(const v16bf*)(wbase + ((16 + 4 * g + p) * 8 + ks) * 512 + lane16);
        accS[p] = wmma_bf16(aS, bop, accS[p]);
      }
    }
  }

  // ---- gate: tanh * sigmoid (channel m pairs with m+256 inside this wave) ----
  v8f acts[4];
#pragma unroll
  for (int p = 0; p < 4; ++p)
#pragma unroll
    for (int r = 0; r < 8; ++r) {
      float zt = accT[p][r], zs = accS[p][r];
      acts[p][r] = tanhf(zt) * (1.0f / (1.0f + __expf(-zs)));
    }

  __syncthreads();   // all waves done reading X before it is overwritten

#pragma unroll
  for (int p = 0; p < 4; ++p) {
    int cb = (4 * g + p) * 16 + 8 * hi;
    union { v8bf v; __bf16 e[8]; } st;
#pragma unroll
    for (int r = 0; r < 8; ++r) st.e[r] = (__bf16)acts[p][r];
    *(v8bf*)(A + nIdx * XP + cb) = st.v;   // acts transposed tile [n][c]
  }
  __syncthreads();

  const bool isFirst = (flags & 1) != 0;
  const bool isLast  = (flags & 2) != 0;

  // ---- res/skip GEMM: 512x256 @ 256x16 ----
  v8f rr[4], rk[4];
#pragma unroll
  for (int p = 0; p < 4; ++p) {
    int mR = (4 * g + p) * 16 + 8 * hi;
    const v4f* br = (const v4f*)(res_b + layer * 512 + mR);
    const v4f* bk = (const v4f*)(res_b + layer * 512 + 256 + mR);
    v4f r0v = br[0], r1v = br[1];
#pragma unroll
    for (int r = 0; r < 4; ++r) { rr[p][r] = r0v[r]; rr[p][r + 4] = r1v[r]; }
    if (isLast) {
#pragma unroll
      for (int r = 0; r < 8; ++r) rk[p][r] = 0.0f;
    } else {
      v4f k0v = bk[0], k1v = bk[1];
#pragma unroll
      for (int r = 0; r < 4; ++r) { rk[p][r] = k0v[r]; rk[p][r + 4] = k1v[r]; }
    }
  }
  {
    const __bf16* rbase = resP + ((long)layer * 32) * 8 * 512;
    const __bf16* arow = A + nIdx * XP + hi * 16;
#pragma unroll
    for (int ks = 0; ks < 8; ++ks) {
      v16bf bop = lds_loadB(arow + ks * 32);
#pragma unroll
      for (int p = 0; p < 4; ++p) {
        v16bf aR = *(const v16bf*)(rbase + ((4 * g + p) * 8 + ks) * 512 + lane16);
        rr[p] = wmma_bf16(aR, bop, rr[p]);
        if (!isLast) {
          v16bf aK = *(const v16bf*)(rbase + ((16 + 4 * g + p) * 8 + ks) * 512 + lane16);
          rk[p] = wmma_bf16(aK, bop, rk[p]);
        }
      }
    }
  }

  // ---- fused writeback: h' = h + rs[:256]; skip accumulation in f32 ----
  int t = t0 + nIdx;
#pragma unroll
  for (int p = 0; p < 4; ++p) {
    int cb = (4 * g + p) * 16 + 8 * hi;
#pragma unroll
    for (int r = 0; r < 8; ++r) {
      long idx = ((long)b * NC + cb + r) * T_LEN + t;
      if (isLast) {
        outAcc[idx] += rr[p][r];               // last layer: rs[:256] -> skip
      } else {
        hOut[idx] = hIn[idx] + rr[p][r];
        outAcc[idx] = isFirst ? rk[p][r] : (outAcc[idx] + rk[p][r]);
      }
    }
  }
}

// ---------------- end conv: 256 -> 16, 1x1 ----------------------------------
__global__ void wn_end(const float* __restrict__ outAcc,
                       const float* __restrict__ ew,
                       const float* __restrict__ eb,
                       float* __restrict__ out) {
  long idx = (long)blockIdx.x * blockDim.x + threadIdx.x;
  if (idx >= (long)NB * T_LEN) return;
  int t = (int)(idx % T_LEN);
  int b = (int)(idx / T_LEN);
  float acc[16];
#pragma unroll
  for (int co = 0; co < 16; ++co) acc[co] = eb[co];
  const float* xb = outAcc + (long)b * NC * T_LEN + t;
  for (int c = 0; c < NC; ++c) {
    float x = xb[(long)c * T_LEN];
#pragma unroll
    for (int co = 0; co < 16; ++co) acc[co] += ew[co * NC + c] * x;
  }
#pragma unroll
  for (int co = 0; co < 16; ++co)
    out[((long)b * 16 + co) * T_LEN + t] = acc[co];
}

extern "C" void kernel_launch(void* const* d_in, const int* in_sizes, int n_in,
                              void* d_out, int out_size, void* d_ws, size_t ws_size,
                              hipStream_t stream) {
  const float* audio   = (const float*)d_in[0];
  const float* spect   = (const float*)d_in[1];
  const float* start_w = (const float*)d_in[2];
  const float* start_b = (const float*)d_in[3];
  const float* cond_w  = (const float*)d_in[4];
  const float* cond_b  = (const float*)d_in[5];
  const float* in_w    = (const float*)d_in[6];
  const float* in_b    = (const float*)d_in[7];
  const float* res_w   = (const float*)d_in[8];
  const float* res_b   = (const float*)d_in[9];
  const float* end_w   = (const float*)d_in[10];
  const float* end_b   = (const float*)d_in[11];
  float* out = (float*)d_out;

  char* ws = (char*)d_ws;
  const size_t HSZ = (size_t)NB * NC * T_LEN * 4;     // 16 MB per h-sized buffer
  float*  h0      = (float*)(ws);
  float*  h1      = (float*)(ws + HSZ);
  float*  outAcc  = (float*)(ws + 2 * HSZ);
  __bf16* inP     = (__bf16*)(ws + 3 * HSZ);
  __bf16* resP    = (__bf16*)(ws + 3 * HSZ + (size_t)NIN_PACK * 2);
  __bf16* condP   = (__bf16*)(ws + 3 * HSZ + ((size_t)NIN_PACK + NRES_PACK) * 2);
  float*  bias512 = (float*)(ws + 3 * HSZ +
                             ((size_t)NIN_PACK + NRES_PACK + NCOND_PACK) * 2);

  {
    long ntot = (long)NIN_PACK + NRES_PACK + NCOND_PACK + NBIAS;
    wn_prep_pack<<<(int)((ntot + 255) / 256), 256, 0, stream>>>(
        in_w, res_w, cond_w, in_b, cond_b, inP, resP, condP, bias512);
  }
  {
    long ntot = (long)NB * NC * T_LEN;
    wn_start<<<(int)((ntot + 255) / 256), 256, 0, stream>>>(audio, start_w,
                                                            start_b, h0);
  }
  float* hbuf[2] = {h0, h1};
  for (int layer = 0; layer < NLAYERS; ++layer) {
    int flags = (layer == 0 ? 1 : 0) | (layer == NLAYERS - 1 ? 2 : 0);
    dim3 grid(T_LEN / TT, NB);
    wn_layer<<<grid, NTHREADS, 0, stream>>>(hbuf[layer & 1], hbuf[(layer + 1) & 1],
                                            outAcc, spect, inP, condP, resP,
                                            bias512, res_b, layer, 1 << layer, flags);
  }
  {
    long ntot = (long)NB * T_LEN;
    wn_end<<<(int)((ntot + 255) / 256), 256, 0, stream>>>(outAcc, end_w, end_b,
                                                          out);
  }
}
